// ChebConvNet_6356551598698
// MI455X (gfx1250) — matrix operand order repaired and checked by reading, not compiled, not run on hardware
//
#include <hip/hip_runtime.h>

// ---------------------------------------------------------------------------
// ChebConv(K=1) net == 3-layer MLP over 500k nodes + row log_softmax.
// FP16 WMMA (v_wmma_f32_16x16x32_f16) with fp32 accumulation.
// Weights live in LDS (padded), activations staged per-wave in LDS.
// ---------------------------------------------------------------------------

#define N_NODES  500000
#define F_IN     128
#define F_HID    128
#define F_OUT    64
#define SROW     136                 // padded LDS row stride (halfs): 272B = 68 banks -> no broadside conflicts
#define TILES_TOTAL (N_NODES / 16)   // 31250, exact

typedef __attribute__((ext_vector_type(16))) _Float16 v16h;
typedef __attribute__((ext_vector_type(8)))  _Float16 v8h;
typedef __attribute__((ext_vector_type(4)))  _Float16 v4h;
typedef __attribute__((ext_vector_type(8)))  float    v8f;
typedef __attribute__((ext_vector_type(4)))  float    v4f;

static __device__ __forceinline__ v16h cat8(v8h lo, v8h hi) {
    return __builtin_shufflevector(lo, hi, 0,1,2,3,4,5,6,7,8,9,10,11,12,13,14,15);
}

// A fragment (16x32 f16, M x K) from the wave's padded LDS activation tile.
// Lane L holds row M = L%16; K = kbase + 8*(L/16) + {0..7} and + {16..23}.
static __device__ __forceinline__ v16h load_a_lds(const _Float16* act, int lane, int kbase) {
    const int row = lane & 15;
    const int sel = (lane >> 4) << 3;              // 0 or 8
    const _Float16* p = act + row * SROW + kbase + sel;
    v8h lo = *(const v8h*)(p);
    v8h hi = *(const v8h*)(p + 16);
    return cat8(lo, hi);
}

// B fragment (32x16 f16, K x N) from transposed LDS weights Wt[n][k] (k contiguous, padded stride).
// Lane L holds col N = 16*ntile + L%16; K = kbase + 8*(L/16) + {0..7} and + {16..23}.
static __device__ __forceinline__ v16h load_b_lds(const _Float16* wt, int lane, int ntile, int kbase) {
    const int col = (ntile << 4) + (lane & 15);
    const int sel = (lane >> 4) << 3;              // 0 or 8
    const _Float16* p = wt + col * SROW + kbase + sel;
    v8h lo = *(const v8h*)(p);
    v8h hi = *(const v8h*)(p + 16);
    return cat8(lo, hi);
}

// One dense layer on a 16-row tile: C[n] = act(16x128) @ W(128 x 16*NT) + bias
template <int NT>
static __device__ __forceinline__ void layer_wmma(const _Float16* act, const _Float16* wt,
                                                  const float* bv, int lane, v8f C[NT]) {
    v16h A[4];
#pragma unroll
    for (int c = 0; c < 4; ++c) A[c] = load_a_lds(act, lane, 32 * c);
#pragma unroll
    for (int n = 0; n < NT; ++n) {
        v8f acc;
#pragma unroll
        for (int i = 0; i < 8; ++i) acc[i] = bv[n];   // bias depends only on N
#pragma unroll
        for (int c = 0; c < 4; ++c) {
            v16h B = load_b_lds(wt, lane, n, 32 * c);
            acc = __builtin_amdgcn_wmma_f32_16x16x32_f16(
                false, A[c], false, B, (short)0, acc, false, false);
        }
        C[n] = acc;
    }
}

// SiLU + repack C-layout fp32 accumulators into the f16 LDS activation tile.
// C element r of lane L sits at (M = r + 8*(L/16), N = 16n + L%16).
template <int NT>
static __device__ __forceinline__ void silu_store(_Float16* act, int lane, const v8f C[NT]) {
    const int mh = (lane >> 4) << 3;   // 0 or 8
    const int nc = lane & 15;
#pragma unroll
    for (int n = 0; n < NT; ++n) {
#pragma unroll
        for (int r = 0; r < 8; ++r) {
            float v = C[n][r];
            float s = v * (1.0f / (1.0f + __expf(-v)));   // SiLU
            act[(r + mh) * SROW + (n << 4) + nc] = (_Float16)s;
        }
    }
}

// ---------------------------------------------------------------------------
// Prep: fp32 weights -> transposed f16 in workspace (unpadded, k contiguous).
// ws layout (halfs, rows of 128): rows [0,128) W0t; [128,256) W1t; [256,320) W2t
// ---------------------------------------------------------------------------
__global__ void prep_weights_kernel(const float* __restrict__ W0,
                                    const float* __restrict__ W1,
                                    const float* __restrict__ W2,
                                    _Float16* __restrict__ ws) {
    int idx = blockIdx.x * blockDim.x + threadIdx.x;
    if (idx < 16384) {
        int n = idx >> 7, k = idx & 127;
        ws[idx] = (_Float16)W0[k * F_HID + n];
    } else if (idx < 32768) {
        int l = idx - 16384;
        int n = l >> 7, k = l & 127;
        ws[idx] = (_Float16)W1[k * F_HID + n];
    } else if (idx < 40960) {
        int l = idx - 32768;
        int n = l >> 7, k = l & 127;
        ws[idx] = (_Float16)W2[k * F_OUT + n];
    }
}

// ---------------------------------------------------------------------------
// Main fused kernel: weights resident in LDS, per-wave 16-row tiles,
// persistent grid-stride loop.
// ---------------------------------------------------------------------------
__global__ void __launch_bounds__(256)
gnn_mlp_kernel(const float* __restrict__ x,
               const _Float16* __restrict__ ws,
               const float* __restrict__ b0,
               const float* __restrict__ b1,
               const float* __restrict__ b2,
               float* __restrict__ out) {
    // 320 weight rows * 136 halfs = 87040 B ; activation tiles 8 * 4352 B = 34816 B
    __shared__ __align__(16) _Float16 wlds[320 * SROW];
    __shared__ __align__(16) _Float16 act_all[8 * 16 * SROW];

    const int lane = threadIdx.x & 31;
    const int wave = threadIdx.x >> 5;
    _Float16* act = act_all + wave * (16 * SROW);

    // ---- Stage transposed f16 weights into padded LDS (coalesced 16B copies) ----
    for (int c = threadIdx.x; c < 320 * 16; c += 256) {
        int row = c >> 4, part = c & 15;
        *(v8h*)(wlds + row * SROW + part * 8) = *(const v8h*)(ws + row * 128 + part * 8);
    }
    __syncthreads();

    // Per-lane bias broadcasts (N = 16n + lane%16), hoisted out of the tile loop.
    const int nc = lane & 15;
    float bv0[8], bv1[8], bv2[4];
#pragma unroll
    for (int n = 0; n < 8; ++n) { bv0[n] = b0[(n << 4) + nc]; bv1[n] = b1[(n << 4) + nc]; }
#pragma unroll
    for (int n = 0; n < 4; ++n) { bv2[n] = b2[(n << 4) + nc]; }

    const int mh = (lane >> 4) << 3;   // row-half offset for C layout

#pragma unroll 1
    for (int t = blockIdx.x * 8 + wave; t < TILES_TOTAL; t += gridDim.x * 8) {
        const int row0 = t << 4;   // 16 node rows per tile

        // Launder the weight base offset so B loads are not provably
        // loop-invariant -> prevents LICM + mass spilling of weight fragments.
        int zoff = 0;
        asm volatile("" : "+v"(zoff));
        const _Float16* W0l = wlds + zoff;                 // rows [0,128)
        const _Float16* W1l = wlds + 128 * SROW + zoff;    // rows [128,256)
        const _Float16* W2l = wlds + 256 * SROW + zoff;    // rows [256,320)

        // ---- Stage x tile [16 x 128] fp32 -> f16 LDS (coalesced float4 loads) ----
        const float* xt = x + (size_t)row0 * F_IN;
#pragma unroll
        for (int i = 0; i < 16; ++i) {
            int f4 = i * 32 + lane;          // 512 float4 per tile
            int r  = f4 >> 5;                // 32 float4 per row
            int c4 = f4 & 31;
            v4f v = *(const v4f*)(xt + r * F_IN + c4 * 4);
            v4h h; h[0] = (_Float16)v[0]; h[1] = (_Float16)v[1];
                   h[2] = (_Float16)v[2]; h[3] = (_Float16)v[3];
            *(v4h*)(act + r * SROW + c4 * 4) = h;
        }

        // ---- Layer 0: [16x128]@[128x128] + b0, SiLU ----
        v8f C8[8];
        layer_wmma<8>(act, W0l, bv0, lane, C8);
        silu_store<8>(act, lane, C8);

        // ---- Layer 1: [16x128]@[128x128] + b1, SiLU ----
        layer_wmma<8>(act, W1l, bv1, lane, C8);
        silu_store<8>(act, lane, C8);

        // ---- Layer 2: [16x128]@[128x64] + b2 ----
        v8f C4[4];
        layer_wmma<4>(act, W2l, bv2, lane, C4);

        // ---- Row-wise log_softmax over 64 outputs + store ----
        // Row M = r + 8*(lane/16) lives in one 16-lane half-wave (4 vals/lane).
#pragma unroll
        for (int r = 0; r < 8; ++r) {
            float m = C4[0][r];
#pragma unroll
            for (int n = 1; n < 4; ++n) m = fmaxf(m, C4[n][r]);
#pragma unroll
            for (int off = 1; off < 16; off <<= 1) m = fmaxf(m, __shfl_xor(m, off, 32));
            float s = 0.0f;
#pragma unroll
            for (int n = 0; n < 4; ++n) s += __expf(C4[n][r] - m);
#pragma unroll
            for (int off = 1; off < 16; off <<= 1) s += __shfl_xor(s, off, 32);
            const float shift = m + __logf(s);
            float* orow = out + (size_t)(row0 + r + mh) * F_OUT;
#pragma unroll
            for (int n = 0; n < 4; ++n) orow[(n << 4) + nc] = C4[n][r] - shift;
        }
    }
}

// ---------------------------------------------------------------------------
// Entry point.
// setup_inputs order: x, edge_index(unused), W0, b0, W1, b1, W2, b2
// ---------------------------------------------------------------------------
extern "C" void kernel_launch(void* const* d_in, const int* in_sizes, int n_in,
                              void* d_out, int out_size, void* d_ws, size_t ws_size,
                              hipStream_t stream) {
    const float* x  = (const float*)d_in[0];
    const float* W0 = (const float*)d_in[2];
    const float* b0 = (const float*)d_in[3];
    const float* W1 = (const float*)d_in[4];
    const float* b1 = (const float*)d_in[5];
    const float* W2 = (const float*)d_in[6];
    const float* b2 = (const float*)d_in[7];
    float* out = (float*)d_out;
    _Float16* wt = (_Float16*)d_ws;   // 40960 halfs = 80 KB

    prep_weights_kernel<<<dim3(160), dim3(256), 0, stream>>>(W0, W1, W2, wt);
    gnn_mlp_kernel<<<dim3(1024), dim3(256), 0, stream>>>(x, wt, b0, b1, b2, out);
}
